// AffinityNeuralNetworkMONN_29300266893467
// MI455X (gfx1250) — matrix operand order repaired and checked by reading, not compiled, not run on hardware
//
#include <hip/hip_runtime.h>
#include <hip/hip_bf16.h>

typedef unsigned short u16b;
typedef __attribute__((ext_vector_type(16))) __bf16 v16bf;
typedef __attribute__((ext_vector_type(8)))  float  v8f;
typedef __attribute__((ext_vector_type(4)))  unsigned int v4u;
typedef __attribute__((ext_vector_type(8)))  int  v8i;
typedef __attribute__((ext_vector_type(4)))  int  v4i;

#define ACT_NONE 0
#define ACT_LRELU 1
#define ACT_TANH 2
#define ACT_SIGMOID 3

#define AS1 __attribute__((address_space(1)))
#define AS3 __attribute__((address_space(3)))
typedef AS1 v4i* gv4i_p;   // global (device) v4i pointer
typedef AS3 v4i* lv4i_p;   // LDS v4i pointer

#if defined(__has_builtin)
#if __has_builtin(__builtin_amdgcn_tensor_load_to_lds) && __has_builtin(__builtin_amdgcn_s_wait_tensorcnt)
#define HAVE_TDM 1
#else
#define HAVE_TDM 0
#endif
#if __has_builtin(__builtin_amdgcn_global_load_async_to_lds_b128)
#define HAVE_ASYNC 1
#else
#define HAVE_ASYNC 0
#endif
#else
#define HAVE_TDM 0
#define HAVE_ASYNC 0
#endif

__device__ __forceinline__ u16b f2bf(float x) {
    unsigned u = __float_as_uint(x);
    unsigned r = (u + 0x7FFFu + ((u >> 16) & 1u)) >> 16;   // RNE
    return (u16b)r;
}

#if HAVE_ASYNC
__device__ __forceinline__ void async_wait0() {
#if __has_builtin(__builtin_amdgcn_s_wait_asynccnt)
    __builtin_amdgcn_s_wait_asynccnt(0);
#else
    asm volatile("s_wait_asynccnt 0x0" ::: "memory");
#endif
}
#endif

#if HAVE_TDM
// 2-D TDM descriptor load: tile [rows][tile_k] bf16, global row stride ld (elems),
// packed into LDS at lds_addr (row stride = tile_k elems).
__device__ __forceinline__ void tdm_load_2d(unsigned lds_addr, const u16b* gptr,
                                            int tile_k, int rows, int ld)
{
    unsigned long long ga = (unsigned long long)(size_t)gptr;
    v4u g0;
    g0[0] = 1u;                                            // count=1 (valid user D#)
    g0[1] = lds_addr;                                      // lds_addr [63:32]
    g0[2] = (unsigned)(ga & 0xFFFFFFFFu);                  // global_addr lo
    g0[3] = (unsigned)(((ga >> 32) & 0x1FFFFFFu) | (2u << 30)); // addr hi | type=2
    unsigned td0 = (unsigned)ld;                           // tensor_dim0
    unsigned td1 = (unsigned)rows;                         // tensor_dim1
    unsigned long long st0 = (unsigned long long)ld;       // tensor_dim0_stride
    v8i g1;
    g1[0] = (int)(1u << 16);                               // wg_mask=0, data_size=1 (2B)
    g1[1] = (int)((td0 & 0xFFFFu) << 16);                  // td0[15:0] @ bits 63:48
    g1[2] = (int)((td0 >> 16) | ((td1 & 0xFFFFu) << 16));  // td0[31:16] | td1[15:0]
    g1[3] = (int)((td1 >> 16) | (((unsigned)tile_k & 0xFFFFu) << 16)); // td1 hi | tile_dim0
    g1[4] = (int)((unsigned)rows & 0xFFFFu);               // tile_dim1 = rows, tile_dim2=0
    g1[5] = (int)(st0 & 0xFFFFFFFFu);                      // dim0_stride lo32
    g1[6] = (int)((st0 >> 32) & 0xFFFFu);                  // dim0_stride hi16, dim1_stride lo=0
    g1[7] = 0;
    v4i z4; z4[0] = 0; z4[1] = 0; z4[2] = 0; z4[3] = 0;
    v8i z8; z8[0] = 0; z8[1] = 0; z8[2] = 0; z8[3] = 0;
    z8[4] = 0; z8[5] = 0; z8[6] = 0; z8[7] = 0;
    __builtin_amdgcn_tensor_load_to_lds(g0, g1, z4, z4, z8, 0);
}
#endif

// ---------------------------------------------------------------------------
// Epilogue helper indices: wave w owns rows [16w,16w+16); acc tile j covers
// cols [16j,16j+16). Element (v,lane): row = 16w + v + 8*(lane>>4), col = lane&15.
// ---------------------------------------------------------------------------

// Full-panel WMMA GEMM (K <= 128): A [M,K] row-major bf16, B stored [N][K] bf16.
// A/B panels for the 64x64 block are loaded whole (TDM when available), then a
// barrier-free ds_load_b128 + v_wmma inner loop.
template<int ACT, bool OF, bool OB, bool OBT>
__global__ __launch_bounds__(128) void gemm_tdm(
    const u16b* __restrict__ A,  long strideA, int lda,
    const u16b* __restrict__ Bm, long strideB, int ldb,
    const float* __restrict__ bias,
    float* __restrict__ Cf, long strideCf,
    u16b*  __restrict__ Cb, long strideCb,
    u16b*  __restrict__ CbT, long strideT, int segT,
    int M, int N, int K, int ldc)
{
    __shared__ u16b lA[64 * 128];
    __shared__ u16b lB[64 * 128];
    const int bz = blockIdx.z;
    A  += (long)bz * strideA;
    Bm += (long)bz * strideB;
    const int m0 = blockIdx.y * 64, n0 = blockIdx.x * 64;
    const int t = threadIdx.x, lane = t & 31, wave = t >> 5;

#if HAVE_TDM
    if (t == 0) {
        tdm_load_2d((unsigned)(size_t)lA, A  + (long)m0 * lda, K, 64, lda);
        tdm_load_2d((unsigned)(size_t)lB, Bm + (long)n0 * ldb, K, 64, ldb);
        __builtin_amdgcn_s_wait_tensorcnt((short)0);
    }
#else
    {
        const int kw = K >> 3;                 // uint4 chunks per row
        for (int c = t; c < 64 * kw; c += 128) {
            int r = c / kw, col = (c - r * kw) * 8;
            *(uint4*)(&lA[r * K + col]) = *(const uint4*)(A + (long)(m0 + r) * lda + col);
        }
        for (int c = t; c < 64 * kw; c += 128) {
            int r = c / kw, col = (c - r * kw) * 8;
            *(uint4*)(&lB[r * K + col]) = *(const uint4*)(Bm + (long)(n0 + r) * ldb + col);
        }
    }
#endif
    __syncthreads();

    v8f acc[4] = {};
    for (int k0 = 0; k0 < K; k0 += 32) {
        union { uint4 u[2]; v16bf v; } fa;
        {
            int row = wave * 16 + (lane & 15);
            int kb  = k0 + (lane >> 4) * 8;
            fa.u[0] = *(const uint4*)(&lA[row * K + kb]);
            fa.u[1] = *(const uint4*)(&lA[row * K + kb + 16]);
        }
        #pragma unroll
        for (int j = 0; j < 4; ++j) {
            union { uint4 u[2]; v16bf v; } fb;
            int nn = j * 16 + (lane & 15);
            int kb = k0 + (lane >> 4) * 8;
            fb.u[0] = *(const uint4*)(&lB[nn * K + kb]);
            fb.u[1] = *(const uint4*)(&lB[nn * K + kb + 16]);
            acc[j] = __builtin_amdgcn_wmma_f32_16x16x32_bf16(
                false, fa.v, false, fb.v, (short)0, acc[j], false, false);
        }
    }

    #pragma unroll
    for (int j = 0; j < 4; ++j) {
        #pragma unroll
        for (int v = 0; v < 8; ++v) {
            int row = m0 + wave * 16 + v + 8 * (lane >> 4);
            int col = n0 + j * 16 + (lane & 15);
            float x = acc[j][v];
            if (bias) x += bias[col];
            if (ACT == ACT_LRELU)   x = (x >= 0.f) ? x : 0.1f * x;
            if (ACT == ACT_TANH)    x = tanhf(x);
            if (ACT == ACT_SIGMOID) x = 1.f / (1.f + __expf(-x));
            if (OF) Cf[(long)bz * strideCf + (long)row * ldc + col] = x;
            if (OB) Cb[(long)bz * strideCb + (long)row * ldc + col] = f2bf(x);
            if (OBT) {
                long tb = (long)bz + row / segT;
                int  tr = row % segT;
                CbT[tb * strideT + (long)col * segT + tr] = f2bf(x);
            }
        }
    }
}

// K-step pipelined WMMA GEMM for large K (p_to_c, K=1024).
// A [M,K] row-major bf16, B stored [N][K] bf16. Async LDS copies when available.
template<int ACT, bool OF>
__global__ __launch_bounds__(128) void gemm_loop(
    const u16b* __restrict__ A,  long strideA, int lda,
    const u16b* __restrict__ Bm, long strideB, int ldb,
    const float* __restrict__ bias,
    float* __restrict__ Cf, long strideCf,
    int M, int N, int K, int ldc)
{
    __shared__ u16b lA[64 * 40];   // [m][k], stride 40 (80B rows, 16B aligned)
    __shared__ u16b lB[64 * 40];   // [n][k]
    const int bz = blockIdx.z;
    A  += (long)bz * strideA;
    Bm += (long)bz * strideB;
    const int m0 = blockIdx.y * 64, n0 = blockIdx.x * 64;
    const int t = threadIdx.x, lane = t & 31, wave = t >> 5;

    v8f acc[4] = {};
    for (int k0 = 0; k0 < K; k0 += 32) {
        __syncthreads();
#if HAVE_ASYNC
        {
            int c = t;
            #pragma unroll
            for (int it = 0; it < 2; ++it, c += 128) {
                int r = c >> 2, col = (c & 3) * 8;
                __builtin_amdgcn_global_load_async_to_lds_b128(
                    (gv4i_p)(size_t)(A + (long)(m0 + r) * lda + k0 + col),
                    (lv4i_p)(unsigned)(size_t)(&lA[r * 40 + col]), 0, 0);
            }
            int n = t >> 1, c0 = (t & 1) * 16;
            __builtin_amdgcn_global_load_async_to_lds_b128(
                (gv4i_p)(size_t)(Bm + (long)(n0 + n) * ldb + k0 + c0),
                (lv4i_p)(unsigned)(size_t)(&lB[n * 40 + c0]), 0, 0);
            __builtin_amdgcn_global_load_async_to_lds_b128(
                (gv4i_p)(size_t)(Bm + (long)(n0 + n) * ldb + k0 + c0 + 8),
                (lv4i_p)(unsigned)(size_t)(&lB[n * 40 + c0 + 8]), 0, 0);
            async_wait0();
        }
#else
        {
            int c = t;
            #pragma unroll
            for (int it = 0; it < 2; ++it, c += 128) {
                int r = c >> 2, col = (c & 3) * 8;
                *(uint4*)(&lA[r * 40 + col]) = *(const uint4*)(A + (long)(m0 + r) * lda + k0 + col);
            }
            int n = t >> 1, c0 = (t & 1) * 16;
            const uint4* src = (const uint4*)(Bm + (long)(n0 + n) * ldb + k0 + c0);
            *(uint4*)(&lB[n * 40 + c0])     = src[0];
            *(uint4*)(&lB[n * 40 + c0 + 8]) = src[1];
        }
#endif
        if (k0 + 32 < K)
            __builtin_prefetch(A + (long)(m0 + (t >> 1)) * lda + k0 + 32, 0, 0);
        __syncthreads();

        union { uint4 u[2]; v16bf v; } fa;
        {
            int row = wave * 16 + (lane & 15);
            int kb  = (lane >> 4) * 8;
            fa.u[0] = *(const uint4*)(&lA[row * 40 + kb]);
            fa.u[1] = *(const uint4*)(&lA[row * 40 + kb + 16]);
        }
        #pragma unroll
        for (int j = 0; j < 4; ++j) {
            union { uint4 u[2]; v16bf v; } fb;
            int nn = j * 16 + (lane & 15);
            int kb = (lane >> 4) * 8;
            fb.u[0] = *(const uint4*)(&lB[nn * 40 + kb]);
            fb.u[1] = *(const uint4*)(&lB[nn * 40 + kb + 16]);
            acc[j] = __builtin_amdgcn_wmma_f32_16x16x32_bf16(
                false, fa.v, false, fb.v, (short)0, acc[j], false, false);
        }
    }

    #pragma unroll
    for (int j = 0; j < 4; ++j) {
        #pragma unroll
        for (int v = 0; v < 8; ++v) {
            int row = m0 + wave * 16 + v + 8 * (lane >> 4);
            int col = n0 + j * 16 + (lane & 15);
            float x = acc[j][v];
            if (bias) x += bias[col];
            if (ACT == ACT_LRELU) x = (x >= 0.f) ? x : 0.1f * x;
            if (ACT == ACT_TANH)  x = tanhf(x);
            if (OF) Cf[(long)bz * strideCf + (long)row * ldc + col] = x;
        }
    }
}

// ---------------------------------------------------------------------------
__global__ void cvt_bf16(const float* __restrict__ x, u16b* __restrict__ y, long n) {
    long i = (long)blockIdx.x * 256 + threadIdx.x;
    if (i < n) y[i] = f2bf(x[i]);
}

// Transpose f32 [H,H] weight slices into bf16 [N][K] layout.
__global__ void cvt_w_t(const float* __restrict__ x, u16b* __restrict__ y, int Hd) {
    __shared__ float tile[16][17];
    int s = blockIdx.z;
    const float* xs = x + (long)s * Hd * Hd;
    u16b* ys = y + (long)s * Hd * Hd;
    int k0 = blockIdx.y * 16, n0 = blockIdx.x * 16;
    int tx = threadIdx.x, ty = threadIdx.y;
    tile[ty][tx] = xs[(long)(k0 + ty) * Hd + n0 + tx];
    __syncthreads();
    ys[(long)(n0 + ty) * Hd + k0 + tx] = f2bf(tile[tx][ty]);
}

// score[row] = dot(g0[row]*mv[row/seg]*msg[row], w1) + b1
__global__ __launch_bounds__(128) void fuse_score(
    const float* __restrict__ g0, const float* __restrict__ mv,
    const float* __restrict__ msg, const float* __restrict__ w1,
    const float* __restrict__ b1, float* __restrict__ score, int segN)
{
    int row = blockIdx.x, h = threadIdx.x;
    int b = row / segN;
    float t = g0[(long)row * 128 + h] * mv[b * 128 + h] * msg[(long)row * 128 + h];
    __shared__ float red[128];
    red[h] = t * w1[h];
    __syncthreads();
    for (int s = 64; s > 0; s >>= 1) { if (h < s) red[h] += red[h + s]; __syncthreads(); }
    if (h == 0) score[row] = red[0] + b1[0];
}

// segment softmax over N then weighted sum of feat -> outf[B,128]
__global__ __launch_bounds__(128) void softmax_wsum(
    const float* __restrict__ score, const float* __restrict__ feat,
    float* __restrict__ outf, int N)
{
    int b = blockIdx.x, h = threadIdx.x;
    const float* s = score + (long)b * N;
    __shared__ float red[128];
    __shared__ float aw[128];
    float mx = -3.4e38f;
    for (int n = h; n < N; n += 128) mx = fmaxf(mx, s[n]);
    red[h] = mx; __syncthreads();
    for (int st = 64; st > 0; st >>= 1) { if (h < st) red[h] = fmaxf(red[h], red[h + st]); __syncthreads(); }
    mx = red[0]; __syncthreads();
    float sm = 0.f;
    for (int n = h; n < N; n += 128) sm += __expf(s[n] - mx);
    red[h] = sm; __syncthreads();
    for (int st = 64; st > 0; st >>= 1) { if (h < st) red[h] += red[h + st]; __syncthreads(); }
    float inv = 1.f / (red[0] + 1e-6f);
    __syncthreads();
    float acc = 0.f;
    const float* fb = feat + (long)b * N * 128;
    for (int n0 = 0; n0 < N; n0 += 128) {
        int chunk = (N - n0 < 128) ? (N - n0) : 128;
        __syncthreads();
        if (h < chunk) aw[h] = __expf(s[n0 + h] - mx) * inv;
        __syncthreads();
        for (int i = 0; i < chunk; ++i) acc += aw[i] * fb[(long)(n0 + i) * 128 + h];
    }
    outf[(long)b * 128 + h] = acc;
}

// m0 = mean_c(ce) * mean_p(pe)
__global__ __launch_bounds__(128) void mean_mul(
    const float* __restrict__ ce, const float* __restrict__ pe, float* __restrict__ m)
{
    int b = blockIdx.x, h = threadIdx.x;
    float sc = 0.f; const float* c = ce + (long)b * 64 * 128;
    for (int n = 0; n < 64; ++n) sc += c[n * 128 + h];
    float sp = 0.f; const float* p = pe + (long)b * 1024 * 128;
    for (int n = 0; n < 1024; ++n) sp += p[n * 128 + h];
    m[b * 128 + h] = (sc / 64.f) * (sp / 1024.f);
}

// GRUCell: m = GRU(cf*pf, m)
__global__ __launch_bounds__(128) void gru_kernel(
    const float* __restrict__ cf, const float* __restrict__ pf, float* __restrict__ m,
    const float* __restrict__ W_ih, const float* __restrict__ b_ih,
    const float* __restrict__ W_hh, const float* __restrict__ b_hh)
{
    int b = blockIdx.x, h = threadIdx.x;
    __shared__ float xs[128], hs[128];
    xs[h] = cf[b * 128 + h] * pf[b * 128 + h];
    hs[h] = m[b * 128 + h];
    __syncthreads();
    float ir = b_ih[h], iz = b_ih[128 + h], in_ = b_ih[256 + h];
    float hr = b_hh[h], hz = b_hh[128 + h], hn  = b_hh[256 + h];
    for (int k = 0; k < 128; ++k) {
        float x = xs[k], hh = hs[k];
        const float* wi = W_ih + k * 384;
        const float* wh = W_hh + k * 384;
        ir += x * wi[h]; iz += x * wi[128 + h]; in_ += x * wi[256 + h];
        hr += hh * wh[h]; hz += hh * wh[128 + h]; hn += hh * wh[256 + h];
    }
    float r = 1.f / (1.f + __expf(-(ir + hr)));
    float z = 1.f / (1.f + __expf(-(iz + hz)));
    float n = tanhf(in_ + r * hn);
    float hv = hs[h];
    __syncthreads();
    m[b * 128 + h] = (1.f - z) * n + z * hv;
}

// out[b] = sum_{i<256,j<128} lrelu(acfeat[i]*pf[j]) * W_out[i*128+j] + b_out
__global__ __launch_bounds__(256) void outhead(
    const float* __restrict__ cf, const float* __restrict__ sf,
    const float* __restrict__ pf, const float* __restrict__ W_out,
    const float* __restrict__ b_out, float* __restrict__ out)
{
    int b = blockIdx.x, i = threadIdx.x;
    __shared__ float pfs[128];
    __shared__ float red[256];
    if (i < 128) pfs[i] = pf[b * 128 + i];
    __syncthreads();
    float a = (i < 128) ? cf[b * 128 + i] : sf[b * 128 + (i - 128)];
    float s = 0.f;
    const float* w = W_out + i * 128;
    for (int j = 0; j < 128; ++j) {
        float k = a * pfs[j];
        k = (k >= 0.f) ? k : 0.1f * k;
        s += k * w[j];
    }
    red[i] = s; __syncthreads();
    for (int st = 128; st > 0; st >>= 1) { if (i < st) red[i] += red[i + st]; __syncthreads(); }
    if (i == 0) out[b] = red[0] + b_out[0];
}

// ---------------------------------------------------------------------------
template<int ACT, bool OF, bool OB, bool OBT>
static void launch_tdm(hipStream_t s,
    const u16b* A, long sA, int lda, const u16b* Bm, long sB, int ldb,
    const float* bias, float* Cf, long sCf, u16b* Cb, long sCb,
    u16b* CbT, long sT, int segT, int M, int N, int K, int ldc, int batches)
{
    dim3 g((unsigned)(N / 64), (unsigned)(M / 64), (unsigned)batches);
    gemm_tdm<ACT, OF, OB, OBT><<<g, dim3(128), 0, s>>>(
        A, sA, lda, Bm, sB, ldb, bias, Cf, sCf, Cb, sCb, CbT, sT, segT, M, N, K, ldc);
}

extern "C" void kernel_launch(void* const* d_in, const int* in_sizes, int n_in,
                              void* d_out, int out_size, void* d_ws, size_t ws_size,
                              hipStream_t stream)
{
    (void)in_sizes; (void)n_in; (void)out_size; (void)ws_size;
    const int B = 128, NC = 64, NP = 1024, H = 128, D = 3;
    const long NCB = (long)B * NC;       // 8192
    const long NPB = (long)B * NP;       // 131072
    const long HH  = (long)H * H;        // 16384

    const float* comp   = (const float*)d_in[0];
    const float* gomp   = (const float*)d_in[1];
    const float* prot   = (const float*)d_in[2];
    const float* W_pc   = (const float*)d_in[5];  const float* b_pc   = (const float*)d_in[6];
    const float* W_pp   = (const float*)d_in[7];  const float* b_pp   = (const float*)d_in[8];
    const float* W_caff = (const float*)d_in[9];  const float* b_caff = (const float*)d_in[10];
    const float* W_paff = (const float*)d_in[11]; const float* b_paff = (const float*)d_in[12];
    const float* W_saff = (const float*)d_in[13]; const float* b_saff = (const float*)d_in[14];
    const float* W_c2p  = (const float*)d_in[15]; const float* b_c2p  = (const float*)d_in[16];
    const float* W_p2c  = (const float*)d_in[17]; const float* b_p2c  = (const float*)d_in[18];
    const float* W_hc0  = (const float*)d_in[19]; const float* b_hc0  = (const float*)d_in[20];
    const float* W_hp0  = (const float*)d_in[21]; const float* b_hp0  = (const float*)d_in[22];
    const float* W_mc1  = (const float*)d_in[23]; const float* b_mc1  = (const float*)d_in[24];
    const float* W_mp1  = (const float*)d_in[25]; const float* b_mp1  = (const float*)d_in[26];
    const float* W_hc1  = (const float*)d_in[27]; const float* b_hc1  = (const float*)d_in[28];
    const float* W_hp1  = (const float*)d_in[29]; const float* b_hp1  = (const float*)d_in[30];
    const float* W_ih   = (const float*)d_in[31]; const float* b_ih   = (const float*)d_in[32];
    const float* W_hh   = (const float*)d_in[33]; const float* b_hh   = (const float*)d_in[34];
    const float* W_out  = (const float*)d_in[35]; const float* b_out  = (const float*)d_in[36];

    char* base = (char*)d_ws; size_t used = 0;
    auto alloc = [&](size_t bytes) -> char* {
        size_t a = (used + 255) & ~(size_t)255; used = a + bytes; return base + a;
    };
    // bf16 scratch
    u16b* comp_b  = (u16b*)alloc(NCB * H * 2);
    u16b* prot_b  = (u16b*)alloc(NPB * H * 2);
    u16b* gomp_b  = (u16b*)alloc(HH * 2);
    u16b* wpcT    = (u16b*)alloc(HH * 2);     // weights pre-transposed to [N][K]
    u16b* wppT    = (u16b*)alloc(HH * 2);
    u16b* wcaT    = (u16b*)alloc(HH * 2);
    u16b* wpaT    = (u16b*)alloc(HH * 2);
    u16b* wsaT    = (u16b*)alloc(HH * 2);
    u16b* wc2pT   = (u16b*)alloc(D * HH * 2);
    u16b* wp2cT   = (u16b*)alloc(D * HH * 2);
    u16b* whc0T   = (u16b*)alloc(D * HH * 2);
    u16b* whp0T   = (u16b*)alloc(D * HH * 2);
    u16b* wmc1T   = (u16b*)alloc(D * HH * 2);
    u16b* wmp1T   = (u16b*)alloc(D * HH * 2);
    u16b* pc_b    = (u16b*)alloc(NCB * H * 2);
    u16b* pp_b    = (u16b*)alloc(NPB * H * 2);
    u16b* ce_b    = (u16b*)alloc(NCB * H * 2);
    u16b* pe_b    = (u16b*)alloc(NPB * H * 2);
    u16b* pw_b    = (u16b*)alloc((long)B * NC * NP * 2);   // [B][NC][NP]
    u16b* pwT_b   = (u16b*)alloc((long)B * NP * NC * 2);   // [B][NP][NC]
    u16b* cpreT_b = (u16b*)alloc((long)B * H * NC * 2);    // [B][H][NC]
    u16b* ppreT_b = (u16b*)alloc((long)B * H * NP * 2);    // [B][H][NP]
    u16b* m_b     = (u16b*)alloc(HH * 2);
    // f32 scratch
    float* ce_f  = (float*)alloc(NCB * H * 4);
    float* pe_f  = (float*)alloc(NPB * H * 4);
    float* c2p_f = (float*)alloc(NPB * H * 4);
    float* p2c_f = (float*)alloc(NCB * H * 4);
    float* hc0_f = (float*)alloc(NCB * H * 4);
    float* hp0_f = (float*)alloc(NPB * H * 4);
    float* mc1_f = (float*)alloc(HH * 4);
    float* mp1_f = (float*)alloc(HH * 4);
    float* sf_f  = (float*)alloc(HH * 4);
    float* m_f   = (float*)alloc(HH * 4);
    float* cf_f  = (float*)alloc(HH * 4);
    float* pf_f  = (float*)alloc(HH * 4);
    float* sc_c  = (float*)alloc(NCB * 4);
    float* sc_p  = (float*)alloc(NPB * 4);

    auto cvt = [&](const float* x, u16b* y, long n) {
        cvt_bf16<<<dim3((unsigned)((n + 255) / 256)), dim3(256), 0, stream>>>(x, y, n);
    };
    auto cvtT = [&](const float* x, u16b* y, int slices) {
        cvt_w_t<<<dim3(H / 16, H / 16, slices), dim3(16, 16), 0, stream>>>(x, y, H);
    };
    // ---- bf16 conversions: activations row-major, weights transposed ----
    cvt(comp, comp_b, NCB * H); cvt(prot, prot_b, NPB * H); cvt(gomp, gomp_b, HH);
    cvtT(W_pc, wpcT, 1); cvtT(W_pp, wppT, 1);
    cvtT(W_caff, wcaT, 1); cvtT(W_paff, wpaT, 1); cvtT(W_saff, wsaT, 1);
    cvtT(W_c2p, wc2pT, D); cvtT(W_p2c, wp2cT, D);
    cvtT(W_hc0, whc0T, D); cvtT(W_hp0, whp0T, D);
    cvtT(W_mc1, wmc1T, D); cvtT(W_mp1, wmp1T, D);

    // ---- projections (lrelu epilogue, full-panel TDM GEMMs) ----
    launch_tdm<ACT_LRELU,false,true,false>(stream, comp_b,0,H, wpcT,0,H, b_pc,
        nullptr,0, pc_b,0, nullptr,0,1, (int)NCB,H,H,H, 1);
    launch_tdm<ACT_LRELU,false,true,false>(stream, prot_b,0,H, wppT,0,H, b_pp,
        nullptr,0, pp_b,0, nullptr,0,1, (int)NPB,H,H,H, 1);
    launch_tdm<ACT_LRELU,true,true,false>(stream, comp_b,0,H, wcaT,0,H, b_caff,
        ce_f,0, ce_b,0, nullptr,0,1, (int)NCB,H,H,H, 1);
    launch_tdm<ACT_LRELU,true,true,false>(stream, prot_b,0,H, wpaT,0,H, b_paff,
        pe_f,0, pe_b,0, nullptr,0,1, (int)NPB,H,H,H, 1);
    launch_tdm<ACT_LRELU,true,false,false>(stream, gomp_b,0,H, wsaT,0,H, b_saff,
        sf_f,0, nullptr,0, nullptr,0,1, B,H,H,H, 1);

    // ---- pairwise = sigmoid(pc @ pp^T): emits [NC][NP] and transposed [NP][NC]
    launch_tdm<ACT_SIGMOID,false,true,true>(stream,
        pc_b,(long)NC*H,H, pp_b,(long)NP*H,H, nullptr,
        nullptr,0, pw_b,(long)NC*NP,
        pwT_b,(long)NP*NC, NC, NC,NP,H,NP, B);

    // ---- m0 = mean(ce)*mean(pe) ----
    mean_mul<<<dim3(B), dim3(H), 0, stream>>>(ce_f, pe_f, m_f);

    for (int d = 0; d < D; ++d) {
        // c_pre / p_pre emitted transposed ([H][seg] per graph) for use as B panels
        launch_tdm<ACT_TANH,false,false,true>(stream, ce_b,0,H, wc2pT+(long)d*HH,0,H, b_c2p+d*H,
            nullptr,0, nullptr,0, cpreT_b,(long)H*NC, NC, (int)NCB,H,H,H, 1);
        launch_tdm<ACT_TANH,false,false,true>(stream, pe_b,0,H, wp2cT+(long)d*HH,0,H, b_p2c+d*H,
            nullptr,0, nullptr,0, ppreT_b,(long)H*NP, NP, (int)NPB,H,H,H, 1);
        launch_tdm<ACT_TANH,true,false,false>(stream, ce_b,0,H, whc0T+(long)d*HH,0,H, b_hc0+d*H,
            hc0_f,0, nullptr,0, nullptr,0,1, (int)NCB,H,H,H, 1);
        launch_tdm<ACT_TANH,true,false,false>(stream, pe_b,0,H, whp0T+(long)d*HH,0,H, b_hp0+d*H,
            hp0_f,0, nullptr,0, nullptr,0,1, (int)NPB,H,H,H, 1);

        cvt(m_f, m_b, HH);
        launch_tdm<ACT_TANH,true,false,false>(stream, m_b,0,H, wmc1T+(long)d*HH,0,H, b_mc1+d*H,
            mc1_f,0, nullptr,0, nullptr,0,1, B,H,H,H, 1);
        launch_tdm<ACT_TANH,true,false,false>(stream, m_b,0,H, wmp1T+(long)d*HH,0,H, b_mp1+d*H,
            mp1_f,0, nullptr,0, nullptr,0,1, B,H,H,H, 1);

        // c_to_p[b] = pairwise[b]^T (NPxNC) @ c_pre[b] (NCxH): A=pwT (row), B=cpreT [H][NC]
        launch_tdm<ACT_NONE,true,false,false>(stream,
            pwT_b,(long)NP*NC,NC, cpreT_b,(long)H*NC,NC, nullptr,
            c2p_f,(long)NP*H, nullptr,0, nullptr,0,1, NP,H,NC,H, B);
        // p_to_c[b] = pairwise[b] (NCxNP) @ p_pre[b] (NPxH): A=pw (row), B=ppreT [H][NP]
        {
            dim3 g(H / 64, NC / 64, B);
            gemm_loop<ACT_NONE,true><<<g, dim3(128), 0, stream>>>(
                pw_b,(long)NC*NP,NP, ppreT_b,(long)H*NP,NP, nullptr,
                p2c_f,(long)NC*H, NC,H,NP,H);
        }

        fuse_score<<<dim3((unsigned)NCB), dim3(H), 0, stream>>>(
            hc0_f, mc1_f, p2c_f, W_hc1 + d * H, b_hc1 + d, sc_c, NC);
        fuse_score<<<dim3((unsigned)NPB), dim3(H), 0, stream>>>(
            hp0_f, mp1_f, c2p_f, W_hp1 + d * H, b_hp1 + d, sc_p, NP);

        softmax_wsum<<<dim3(B), dim3(H), 0, stream>>>(sc_c, ce_f, cf_f, NC);
        softmax_wsum<<<dim3(B), dim3(H), 0, stream>>>(sc_p, pe_f, pf_f, NP);

        gru_kernel<<<dim3(B), dim3(H), 0, stream>>>(cf_f, pf_f, m_f, W_ih, b_ih, W_hh, b_hh);
    }

    outhead<<<dim3(B), dim3(256), 0, stream>>>(cf_f, sf_f, pf_f, W_out, b_out, (float*)d_out);
}
